// LLM_mlh_Attention_53635551592830
// MI455X (gfx1250) — matrix operand order repaired and checked by reading, not compiled, not run on hardware
//
#include <hip/hip_runtime.h>

// ---------------- problem constants ----------------
#define B_      4
#define S_      4096
#define DM      768
#define HEADS_  8
#define DH_     96
#define NOPE_   48
#define ROPE_   48
#define BLOCK_  256
#define SAMPLE_ 256
#define QPROJ   384
#define KVPROJ  512
#define NPROJ   7
#define NB_     (S_/BLOCK_)      // 16
#define BS_     (B_*S_)          // 16384
#define BH_     (B_*HEADS_)      // 32
#define CKVW    (KVPROJ + ROPE_) // 560 (logical)
#define CKVP    576              // 560 padded to 64-multiple (GEMM N + row stride)
#define KVW     (DM + HEADS_*NOPE_) // 1152

typedef _Float16 f16;
typedef __attribute__((ext_vector_type(16))) _Float16 v16h;
typedef __attribute__((ext_vector_type(8)))  _Float16 v8h;
typedef __attribute__((ext_vector_type(8)))  float    v8f;

static __device__ __forceinline__ v8f zero8() {
  v8f r;
#pragma unroll
  for (int i = 0; i < 8; ++i) r[i] = 0.0f;
  return r;
}

static __device__ __forceinline__ v8f wmma32(v16h a, v16h b, v8f c) {
  // D = A(16x32 f16) x B(32x16 f16) + C(16x16 f32)
  return __builtin_amdgcn_wmma_f32_16x16x32_f16(false, a, false, b, (short)0, c, false, false);
}

// A operand: row-major [16 x >=K] storage; lane l&15 -> row; K chunks {hb..hb+7, 16+hb..}
static __device__ __forceinline__ v16h load_frag_a(const f16* base, int rowStride, int kbase, int lane) {
  int r = lane & 15, hb = (lane >> 4) * 8;
  const f16* p = base + (size_t)r * rowStride + kbase + hb;
  v16h f;
  ((v8h*)&f)[0] = *(const v8h*)(p);
  ((v8h*)&f)[1] = *(const v8h*)(p + 16);
  return f;
}
// B operand from Bt (N x K row-major): lane l&15 -> column n of B; K chunks {hb..hb+7, 16+hb..}
static __device__ __forceinline__ v16h load_frag_b(const f16* baseT, int rowStride, int kbase, int lane) {
  int n = lane & 15, hb = (lane >> 4) * 8;
  const f16* p = baseT + (size_t)n * rowStride + kbase + hb;
  v16h f;
  ((v8h*)&f)[0] = *(const v8h*)(p);
  ((v8h*)&f)[1] = *(const v8h*)(p + 16);
  return f;
}

static __device__ __forceinline__ float redmax16(float x) {
  x = fmaxf(x, __shfl_xor(x, 1, 16));
  x = fmaxf(x, __shfl_xor(x, 2, 16));
  x = fmaxf(x, __shfl_xor(x, 4, 16));
  x = fmaxf(x, __shfl_xor(x, 8, 16));
  return x;
}
static __device__ __forceinline__ float redsum16(float x) {
  x += __shfl_xor(x, 1, 16);
  x += __shfl_xor(x, 2, 16);
  x += __shfl_xor(x, 4, 16);
  x += __shfl_xor(x, 8, 16);
  return x;
}

// ---------------- weight pre-pack: W -> Bt[Npad][K] f16 (transposed, zero-padded) ----------------
__global__ __launch_bounds__(256)
void k_packw(const float* __restrict__ W, f16* __restrict__ Bt,
             int K, int N, int alreadyT)
{
  int idx = blockIdx.x * 256 + threadIdx.x;  // n * K + k, n < Npad
  int n = idx / K, k = idx - n * K;
  float v = 0.0f;
  if (n < N) v = alreadyT ? W[(size_t)n * K + k] : W[(size_t)k * N + n];
  Bt[idx] = (f16)v;
}

// ---------------- WMMA GEMM: C(MxN,f32) = A(MxK,f32) * B, with B given as Bt[N][K] f16 ----------
// M % 128 == 0, N % 64 == 0, K % 32 == 0.  128x64 tile per WG (8 waves, 2x2 WMMA tiles each).
// B tile staged via CDNA5 async global->LDS copy (no VGPR round-trip).
__global__ __launch_bounds__(256)
void k_gemm(const float* __restrict__ A, const f16* __restrict__ Bt,
            float* __restrict__ C, int M, int N, int K)
{
  __shared__ f16 As[128 * 32];
  __shared__ f16 Bs[64 * 32];
  const int t = threadIdx.x;
  const int lane = t & 31;
  const int wave = t >> 5;
  const int m0 = blockIdx.x * 128;
  const int n0 = blockIdx.y * 64;
  const int rw = (wave & 3) * 32;   // wave's 32-row strip
  const int cw = (wave >> 2) * 32;  // wave's 32-col strip
  const int ar = t >> 1, ak = (t & 1) * 16;   // A staging: 2 threads per row, 16 f32 each
  const int bn = t >> 2, bk = (t & 3) * 8;    // B staging: 4 threads per row, 8 f16 each
  const unsigned bsAddr = (unsigned)(size_t)(&Bs[bn * 32 + bk]);  // LDS byte offset (low 32 bits)
  v8f acc[2][2];
#pragma unroll
  for (int i = 0; i < 2; ++i)
#pragma unroll
    for (int j = 0; j < 2; ++j) acc[i][j] = zero8();

  for (int k0 = 0; k0 < K; k0 += 32) {
    __syncthreads();
    // ---- stage A tile (f32 -> f16) ----
    {
      const float* sa = A + (size_t)(m0 + ar) * K + k0 + ak;
      const float4* s4 = (const float4*)sa;
      float4 x0 = s4[0], x1 = s4[1], x2 = s4[2], x3 = s4[3];
      v16h hv;
      hv[0] = (f16)x0.x; hv[1] = (f16)x0.y; hv[2] = (f16)x0.z; hv[3] = (f16)x0.w;
      hv[4] = (f16)x1.x; hv[5] = (f16)x1.y; hv[6] = (f16)x1.z; hv[7] = (f16)x1.w;
      hv[8] = (f16)x2.x; hv[9] = (f16)x2.y; hv[10] = (f16)x2.z; hv[11] = (f16)x2.w;
      hv[12] = (f16)x3.x; hv[13] = (f16)x3.y; hv[14] = (f16)x3.z; hv[15] = (f16)x3.w;
      *(v16h*)(&As[ar * 32 + ak]) = hv;
      if (k0 + 32 < K) __builtin_prefetch(sa + 32, 0, 0);  // global_prefetch_b8
    }
    // ---- stage B tile: async copy 16B/lane straight into LDS ----
    {
      unsigned long long ga =
          (unsigned long long)(const void*)(Bt + (size_t)(n0 + bn) * K + k0 + bk);
      asm volatile("global_load_async_to_lds_b128 %0, %1, off"
                   :: "v"(bsAddr), "v"(ga) : "memory");
      asm volatile("s_wait_asynccnt 0x0" ::: "memory");
    }
    __syncthreads();
    // ---- 2x2 WMMA tiles per wave ----
    v16h a0 = load_frag_a(As + rw * 32, 32, 0, lane);
    v16h a1 = load_frag_a(As + (rw + 16) * 32, 32, 0, lane);
    v16h b0 = load_frag_b(Bs + cw * 32, 32, 0, lane);
    v16h b1 = load_frag_b(Bs + (cw + 16) * 32, 32, 0, lane);
    acc[0][0] = wmma32(a0, b0, acc[0][0]);
    acc[0][1] = wmma32(a0, b1, acc[0][1]);
    acc[1][0] = wmma32(a1, b0, acc[1][0]);
    acc[1][1] = wmma32(a1, b1, acc[1][1]);
  }
  const int hlf = lane >> 4, nl = lane & 15;
#pragma unroll
  for (int i = 0; i < 2; ++i) {
#pragma unroll
    for (int v = 0; v < 8; ++v) {
      int row = m0 + rw + i * 16 + v + 8 * hlf;
#pragma unroll
      for (int j = 0; j < 2; ++j) {
        int col = n0 + cw + j * 16 + nl;
        C[(size_t)row * N + col] = acc[i][j][v];
      }
    }
  }
}

// ---------------- layernorm over L cols of a row (src stride may differ) ----------------
__global__ __launch_bounds__(256)
void k_layernorm(const float* __restrict__ X, int srcStride, int L,
                 const float* __restrict__ g, const float* __restrict__ bt,
                 float* __restrict__ Y)
{
  __shared__ float red[18];
  int row = blockIdx.x;
  const float* x = X + (size_t)row * srcStride;
  float s = 0.f, ss = 0.f;
  for (int i = threadIdx.x; i < L; i += 256) { float v = x[i]; s += v; ss += v * v; }
#pragma unroll
  for (int m = 16; m; m >>= 1) { s += __shfl_xor(s, m, 32); ss += __shfl_xor(ss, m, 32); }
  int wave = threadIdx.x >> 5, lane = threadIdx.x & 31;
  if (lane == 0) { red[wave] = s; red[9 + wave] = ss; }
  __syncthreads();
  if (threadIdx.x == 0) {
    float a = 0.f, b2 = 0.f;
    for (int w = 0; w < 8; ++w) { a += red[w]; b2 += red[9 + w]; }
    float mean = a / (float)L;
    float var = b2 / (float)L - mean * mean;
    red[16] = mean;
    red[17] = rsqrtf(var + 1e-5f);
  }
  __syncthreads();
  float mean = red[16], rstd = red[17];
  float* y = Y + (size_t)row * L;
  for (int i = threadIdx.x; i < L; i += 256)
    y[i] = (x[i] - mean) * rstd * g[i] + bt[i];
}

// ---------------- RoPE + assemble q,k,v (b,h,s,96) ----------------
__global__ __launch_bounds__(256)
void k_assemble(const float* __restrict__ Qbuf, const float* __restrict__ KVbuf,
                const float* __restrict__ ckv,
                float* __restrict__ q, float* __restrict__ k, float* __restrict__ v)
{
  int idx = blockIdx.x * 256 + threadIdx.x;  // (b*S+s)*H + h
  int h = idx % HEADS_;
  int bs = idx / HEADS_;
  int s = bs % S_;
  int b = bs / S_;
  float cs[24], sn[24];
#pragma unroll
  for (int i = 0; i < 24; ++i) {
    float freq = __expf(-(float)i * (9.210340371976184f / 48.0f)); // 10000^(-i/48)
    __sincosf((float)s * freq, &sn[i], &cs[i]);
  }
  size_t orow = (((size_t)b * HEADS_ + h) * S_ + s) * DH_;
  {
    const float* src = Qbuf + (size_t)bs * DM + h * DH_;
    float* dst = q + orow;
    for (int d = 0; d < NOPE_; ++d) dst[d] = src[d];
    const float* r = src + NOPE_;
    for (int j = 0; j < ROPE_; ++j) {
      int jm = (j < 24) ? j : j - 24;
      float rh = (j < 24) ? -r[j + 24] : r[j - 24];
      dst[NOPE_ + j] = r[j] * cs[jm] + rh * sn[jm];
    }
  }
  {
    const float* kn = KVbuf + (size_t)bs * KVW + h * (DH_ + NOPE_);
    const float* kr = ckv + (size_t)bs * CKVP + KVPROJ;
    float* dst = k + orow;
    for (int d = 0; d < NOPE_; ++d) dst[d] = kn[d];
    for (int j = 0; j < ROPE_; ++j) {
      int jm = (j < 24) ? j : j - 24;
      float rh = (j < 24) ? -kr[j + 24] : kr[j - 24];
      dst[NOPE_ + j] = kr[j] * cs[jm] + rh * sn[jm];
    }
  }
  {
    const float* vv = KVbuf + (size_t)bs * KVW + h * (DH_ + NOPE_) + NOPE_;
    float* dst = v + orow;
    for (int d = 0; d < DH_; ++d) dst[d] = vv[d];
  }
}

// ---------------- LSH bucket keys ----------------
__global__ __launch_bounds__(256)
void k_lsh(const float* __restrict__ vecs, const float* __restrict__ pd,
           const int* __restrict__ perm, int* __restrict__ keys)
{
  int idx = blockIdx.x * 256 + threadIdx.x;  // bh*S + s
  const float* x = vecs + (size_t)idx * DH_;
  float acc[NPROJ] = {0.f, 0.f, 0.f, 0.f, 0.f, 0.f, 0.f};
  for (int d = 0; d < DH_; ++d) {
    float xv = x[d];
#pragma unroll
    for (int j = 0; j < NPROJ; ++j) acc[j] += xv * pd[d * NPROJ + j];
  }
  int id = 0;
#pragma unroll
  for (int j = 0; j < NPROJ; ++j) if (acc[j] > 0.f) id |= (1 << j);
  keys[idx] = perm[id];
}

// ---------------- stable counting sort (128 buckets) per (b,h) ----------------
__global__ __launch_bounds__(128)
void k_sort(const int* __restrict__ keys, int* __restrict__ idxOut)
{
  __shared__ unsigned char sk[S_];
  __shared__ int cnt[128];
  __shared__ int start[128];
  int bh = blockIdx.x;
  const int* kk = keys + (size_t)bh * S_;
  cnt[threadIdx.x] = 0;
  __syncthreads();
  for (int i = threadIdx.x; i < S_; i += 128) {
    int kv = kk[i];
    sk[i] = (unsigned char)kv;
    atomicAdd(&cnt[kv], 1);
  }
  __syncthreads();
  if (threadIdx.x == 0) {
    int acc = 0;
    for (int b = 0; b < 128; ++b) { start[b] = acc; acc += cnt[b]; }
  }
  __syncthreads();
  int bucket = threadIdx.x;
  int pos = start[bucket];
  int* out = idxOut + (size_t)bh * S_;
  for (int i = 0; i < S_; ++i)
    if (sk[i] == (unsigned char)bucket) out[pos++] = i;
}

// ---------------- gather sorted qs/ks (f16, row) + vsT (f16, transposed) ----------------
__global__ __launch_bounds__(256)
void k_gather(const float* __restrict__ q, const float* __restrict__ k, const float* __restrict__ v,
              const int* __restrict__ qIdx, const int* __restrict__ kIdx,
              f16* __restrict__ qs, f16* __restrict__ ks, f16* __restrict__ vsT)
{
  int idx = blockIdx.x * 256 + threadIdx.x;  // bh*S + n
  int bh = idx / S_;
  int n = idx % S_;
  int qi = qIdx[idx];
  int ki = kIdx[idx];
  const float* qsrc = q + ((size_t)bh * S_ + qi) * DH_;
  const float* ksrc = k + ((size_t)bh * S_ + ki) * DH_;
  const float* vsrc = v + ((size_t)bh * S_ + ki) * DH_;
  f16* qdst = qs + ((size_t)bh * S_ + n) * DH_;
  f16* kdst = ks + ((size_t)bh * S_ + n) * DH_;
  f16* vt = vsT + (size_t)bh * DH_ * S_ + n;
  for (int d = 0; d < DH_; ++d) {
    qdst[d] = (f16)qsrc[d];
    kdst[d] = (f16)ksrc[d];
    vt[(size_t)d * S_] = (f16)vsrc[d];
  }
}

// ---------------- gather sampled ksub / vsubT ----------------
__global__ __launch_bounds__(256)
void k_gather_sampled(const f16* __restrict__ ks, const f16* __restrict__ vsT,
                      const int* __restrict__ samp,
                      f16* __restrict__ ksub, f16* __restrict__ vsubT)
{
  int idx = blockIdx.x * 256 + threadIdx.x;  // bh*SAMPLE + j
  int bh = idx / SAMPLE_;
  int j = idx % SAMPLE_;
  int sidx = samp[idx];
  const f16* src = ks + ((size_t)bh * S_ + sidx) * DH_;
  f16* dst = ksub + ((size_t)bh * SAMPLE_ + j) * DH_;
  for (int d = 0; d < DH_; ++d) dst[d] = src[d];
  const f16* vt = vsT + (size_t)bh * DH_ * S_ + sidx;
  f16* vo = vsubT + (size_t)bh * DH_ * SAMPLE_ + j;
  for (int d = 0; d < DH_; ++d) vo[(size_t)d * SAMPLE_] = vt[(size_t)d * S_];
}

// ---------------- block-diagonal attention (sorted domain) ----------------
// One WG per (b,h,block): 256 q rows x 256 keys, K=96.  Writes Xf (flat (h,n,q,d) per batch) + lse.
__global__ __launch_bounds__(256)
void k_blockattn(const f16* __restrict__ qs, const f16* __restrict__ ks,
                 const f16* __restrict__ vsT,
                 float* __restrict__ Xf, float* __restrict__ lseB)
{
  __shared__ float sM[256], sS1[256], sS2[256];
  __shared__ f16 Pst[8][32 * 32];
  const int t = threadIdx.x, lane = t & 31, wave = t >> 5;
  const int hlf = lane >> 4, nl = lane & 15;
  const int blk = blockIdx.x;
  const int n = blk % NB_;
  const int bh = blk / NB_;
  const int b = bh / HEADS_, h = bh % HEADS_;
  const f16* Qb = qs + ((size_t)bh * S_ + n * BLOCK_) * DH_;
  const f16* Kb = ks + ((size_t)bh * S_ + n * BLOCK_) * DH_;
  const f16* Vt = vsT + (size_t)bh * DH_ * S_;
  const int rbase = wave * 32;
  const float scale1 = 0.10206207261596575f;  // 96^-0.5  (softmax)
  const float scale2 = 0.03608439182435161f;  // 768^-0.5 (lse)
  sM[rbase + lane] = -3.0e38f;
  sS1[rbase + lane] = 0.f;
  sS2[rbase + lane] = 0.f;
  v16h Qf[2][3];
#pragma unroll
  for (int rt = 0; rt < 2; ++rt)
#pragma unroll
    for (int kc = 0; kc < 3; ++kc)
      Qf[rt][kc] = load_frag_a(Qb + (size_t)(rbase + rt * 16) * DH_, DH_, kc * 32, lane);
  // pass 1: raw row max
  for (int ct = 0; ct < 16; ++ct) {
    const f16* Kt = Kb + (size_t)(ct * 16) * DH_;
    v16h Kf0 = load_frag_b(Kt, DH_, 0, lane);
    v16h Kf1 = load_frag_b(Kt, DH_, 32, lane);
    v16h Kf2 = load_frag_b(Kt, DH_, 64, lane);
#pragma unroll
    for (int rt = 0; rt < 2; ++rt) {
      v8f s = zero8();
      s = wmma32(Qf[rt][0], Kf0, s);
      s = wmma32(Qf[rt][1], Kf1, s);
      s = wmma32(Qf[rt][2], Kf2, s);
#pragma unroll
      for (int v = 0; v < 8; ++v) {
        float m = redmax16(s[v]);
        if (nl == 0) {
          int r = rbase + rt * 16 + v + 8 * hlf;
          sM[r] = fmaxf(sM[r], m);
        }
      }
    }
  }
  // pass 2: recompute, exp, accumulate sums, P@V
  v8f O[2][6];
#pragma unroll
  for (int rt = 0; rt < 2; ++rt)
#pragma unroll
    for (int nt = 0; nt < 6; ++nt) O[rt][nt] = zero8();
  f16* Pw = Pst[wave];
  for (int c2 = 0; c2 < 8; ++c2) {
#pragma unroll
    for (int sub = 0; sub < 2; ++sub) {
      int ct = c2 * 2 + sub;
      const f16* Kt = Kb + (size_t)(ct * 16) * DH_;
      v16h Kf0 = load_frag_b(Kt, DH_, 0, lane);
      v16h Kf1 = load_frag_b(Kt, DH_, 32, lane);
      v16h Kf2 = load_frag_b(Kt, DH_, 64, lane);
#pragma unroll
      for (int rt = 0; rt < 2; ++rt) {
        v8f s = zero8();
        s = wmma32(Qf[rt][0], Kf0, s);
        s = wmma32(Qf[rt][1], Kf1, s);
        s = wmma32(Qf[rt][2], Kf2, s);
#pragma unroll
        for (int v = 0; v < 8; ++v) {
          int r = rbase + rt * 16 + v + 8 * hlf;
          float M = sM[r];
          float p1 = __expf(scale1 * (s[v] - M));
          float p2 = __expf(scale2 * (s[v] - M));
          float r1 = redsum16(p1);
          float r2 = redsum16(p2);
          if (nl == 0) { sS1[r] += r1; sS2[r] += r2; }
          Pw[(rt * 16 + v + 8 * hlf) * 32 + sub * 16 + nl] = (f16)p1;
        }
      }
    }
#pragma unroll
    for (int rt = 0; rt < 2; ++rt) {
      v16h Af = load_frag_a(Pw + rt * 16 * 32, 32, 0, lane);
#pragma unroll
      for (int nt = 0; nt < 6; ++nt) {
        v16h Bf = load_frag_b(Vt + (size_t)(nt * 16) * S_ + n * BLOCK_ + c2 * 32, S_, 0, lane);
        O[rt][nt] = wmma32(Af, Bf, O[rt][nt]);
      }
    }
  }
  // epilogue: normalized output into Xf flat layout  ((h*16+n)*256+q)*96+d
  float* Xb = Xf + (size_t)b * ((size_t)S_ * DM) + (size_t)(h * NB_ + n) * BLOCK_ * DH_;
#pragma unroll
  for (int rt = 0; rt < 2; ++rt) {
#pragma unroll
    for (int v = 0; v < 8; ++v) {
      int r = rbase + rt * 16 + v + 8 * hlf;
      float inv = 1.0f / sS1[r];
#pragma unroll
      for (int nt = 0; nt < 6; ++nt) {
        int d = nt * 16 + nl;
        Xb[(size_t)r * DH_ + d] = O[rt][nt][v] * inv;
      }
    }
  }
  {
    int r = rbase + lane;
    lseB[(size_t)bh * S_ + n * BLOCK_ + r] = scale2 * sM[r] + __logf(sS2[r]);
  }
}

// ---------------- sampled residual attention + LSE blend + un-sort scatter ----------------
__global__ __launch_bounds__(256)
void k_sampledattn(const f16* __restrict__ qs, const f16* __restrict__ ksub,
                   const f16* __restrict__ vsubT,
                   const int* __restrict__ samp, const int* __restrict__ qIdx,
                   const float* __restrict__ lseB, const float* __restrict__ Y,
                   float* __restrict__ out)
{
  __shared__ float sM[256], sS[256];
  __shared__ float colBias[256];
  __shared__ int rowOrig[256];
  __shared__ f16 Pst[8][32 * 32];
  const int t = threadIdx.x, lane = t & 31, wave = t >> 5;
  const int hlf = lane >> 4, nl = lane & 15;
  const int blk = blockIdx.x;
  const int chunk = blk % NB_;  // q chunk == q block id
  const int bh = blk / NB_;
  const int b = bh / HEADS_, h = bh % HEADS_;
  const float scale = 0.03608439182435161f;  // 768^-0.5
  const float LOG16 = 2.772588722239781f;    // log(S/SAMPLE)
  {
    int sp = samp[(size_t)bh * SAMPLE_ + t];
    colBias[t] = ((sp >> 8) == chunk) ? -1.0e30f : 0.0f;
    rowOrig[t] = qIdx[(size_t)bh * S_ + chunk * BLOCK_ + t];
  }
  const int rbase = wave * 32;
  sM[rbase + lane] = -3.0e38f;
  sS[rbase + lane] = 0.f;
  __syncthreads();
  const f16* Qb = qs + ((size_t)bh * S_ + chunk * BLOCK_) * DH_;
  const f16* Kb = ksub + (size_t)bh * SAMPLE_ * DH_;
  const f16* Vt = vsubT + (size_t)bh * DH_ * SAMPLE_;
  v16h Qf[2][3];
#pragma unroll
  for (int rt = 0; rt < 2; ++rt)
#pragma unroll
    for (int kc = 0; kc < 3; ++kc)
      Qf[rt][kc] = load_frag_a(Qb + (size_t)(rbase + rt * 16) * DH_, DH_, kc * 32, lane);
  // pass 1: max of scaled+biased scores
  for (int ct = 0; ct < 16; ++ct) {
    const f16* Kt = Kb + (size_t)(ct * 16) * DH_;
    v16h Kf0 = load_frag_b(Kt, DH_, 0, lane);
    v16h Kf1 = load_frag_b(Kt, DH_, 32, lane);
    v16h Kf2 = load_frag_b(Kt, DH_, 64, lane);
    float cb = colBias[ct * 16 + nl];
#pragma unroll
    for (int rt = 0; rt < 2; ++rt) {
      v8f s = zero8();
      s = wmma32(Qf[rt][0], Kf0, s);
      s = wmma32(Qf[rt][1], Kf1, s);
      s = wmma32(Qf[rt][2], Kf2, s);
#pragma unroll
      for (int v = 0; v < 8; ++v) {
        float m = redmax16(scale * s[v] + cb);
        if (nl == 0) {
          int r = rbase + rt * 16 + v + 8 * hlf;
          sM[r] = fmaxf(sM[r], m);
        }
      }
    }
  }
  // pass 2
  v8f O[2][6];
#pragma unroll
  for (int rt = 0; rt < 2; ++rt)
#pragma unroll
    for (int nt = 0; nt < 6; ++nt) O[rt][nt] = zero8();
  f16* Pw = Pst[wave];
  for (int c2 = 0; c2 < 8; ++c2) {
#pragma unroll
    for (int sub = 0; sub < 2; ++sub) {
      int ct = c2 * 2 + sub;
      const f16* Kt = Kb + (size_t)(ct * 16) * DH_;
      v16h Kf0 = load_frag_b(Kt, DH_, 0, lane);
      v16h Kf1 = load_frag_b(Kt, DH_, 32, lane);
      v16h Kf2 = load_frag_b(Kt, DH_, 64, lane);
      float cb = colBias[ct * 16 + nl];
#pragma unroll
      for (int rt = 0; rt < 2; ++rt) {
        v8f s = zero8();
        s = wmma32(Qf[rt][0], Kf0, s);
        s = wmma32(Qf[rt][1], Kf1, s);
        s = wmma32(Qf[rt][2], Kf2, s);
#pragma unroll
        for (int v = 0; v < 8; ++v) {
          int r = rbase + rt * 16 + v + 8 * hlf;
          float p = __expf(scale * s[v] + cb - sM[r]);
          float rs = redsum16(p);
          if (nl == 0) sS[r] += rs;
          Pw[(rt * 16 + v + 8 * hlf) * 32 + sub * 16 + nl] = (f16)p;
        }
      }
    }
#pragma unroll
    for (int rt = 0; rt < 2; ++rt) {
      v16h Af = load_frag_a(Pw + rt * 16 * 32, 32, 0, lane);
#pragma unroll
      for (int nt = 0; nt < 6; ++nt) {
        v16h Bf = load_frag_b(Vt + (size_t)(nt * 16) * SAMPLE_ + c2 * 32, SAMPLE_, 0, lane);
        O[rt][nt] = wmma32(Af, Bf, O[rt][nt]);
      }
    }
  }
  // epilogue: lse blend with block path + scatter to unsorted output
  const float* Yb = Y + (size_t)b * ((size_t)S_ * DM) + (size_t)h * ((size_t)S_ * DH_);
#pragma unroll
  for (int rt = 0; rt < 2; ++rt) {
#pragma unroll
    for (int v = 0; v < 8; ++v) {
      int r = rbase + rt * 16 + v + 8 * hlf;
      int srow = chunk * BLOCK_ + r;
      float lr = sM[r] + __logf(sS[r]) + LOG16;
      float lb = lseB[(size_t)bh * S_ + srow];
      float c = 1.0f / (1.0f + __expf(lr - lb));
      float invs = 1.0f / sS[r];
      int orig = rowOrig[r];
      float* orow = out + ((size_t)b * S_ + orig) * DM + h * DH_;
#pragma unroll
      for (int nt = 0; nt < 6; ++nt) {
        int d = nt * 16 + nl;
        float blockV = Yb[(size_t)srow * DH_ + d];
        float resV = O[rt][nt][v] * invs;
        orow[d] = c * blockV + (1.0f - c) * resV;
      }
    }
  }
}

// ---------------- host orchestration ----------------
extern "C" void kernel_launch(void* const* d_in, const int* in_sizes, int n_in,
                              void* d_out, int out_size, void* d_ws, size_t ws_size,
                              hipStream_t stream)
{
  (void)in_sizes; (void)n_in; (void)out_size; (void)ws_size;
  const float* x     = (const float*)d_in[0];
  const float* W_dq  = (const float*)d_in[1];
  const float* W_uq  = (const float*)d_in[2];
  const float* g_q   = (const float*)d_in[3];
  const float* b_q   = (const float*)d_in[4];
  const float* W_dkv = (const float*)d_in[5];
  const float* g_kv  = (const float*)d_in[6];
  const float* b_kv  = (const float*)d_in[7];
  const float* W_ukv = (const float*)d_in[8];
  const float* W_o   = (const float*)d_in[9];
  const float* pd    = (const float*)d_in[10];
  const int*   samp  = (const int*)d_in[11];
  const int*   perm  = (const int*)d_in[12];
  float* out = (float*)d_out;

  char* ws = (char*)d_ws;
  size_t off = 0;
  auto alloc = [&](size_t bytes) -> char* {
    char* p = ws + off;
    off += (bytes + 255) & ~(size_t)255;
    return p;
  };
  const size_t BHS = (size_t)B_ * HEADS_ * S_;
  // persistent across both phases
  float* qv = (float*)alloc(BHS * DH_ * 4);
  float* kv = (float*)alloc(BHS * DH_ * 4);
  float* vv = (float*)alloc(BHS * DH_ * 4);
  f16* Bt_dq  = (f16*)alloc((size_t)QPROJ * DM * 2);    // 384 x 768
  f16* Bt_uq  = (f16*)alloc((size_t)DM * QPROJ * 2);    // 768 x 384
  f16* Bt_dkv = (f16*)alloc((size_t)CKVP * DM * 2);     // 576 x 768
  f16* Bt_ukv = (f16*)alloc((size_t)KVW * KVPROJ * 2);  // 1152 x 512
  f16* Bt_o   = (f16*)alloc((size_t)DM * DM * 2);       // 768 x 768
  size_t phase = off;
  // phase A: projections (dead after k_assemble)
  float* tmp1 = (float*)alloc((size_t)BS_ * QPROJ * 4);
  float* cq   = (float*)alloc((size_t)BS_ * QPROJ * 4);
  float* Qbuf = (float*)alloc((size_t)BS_ * DM * 4);
  float* ckv  = (float*)alloc((size_t)BS_ * CKVP * 4);
  float* kvl  = (float*)alloc((size_t)BS_ * KVPROJ * 4);
  float* KVb  = (float*)alloc((size_t)BS_ * KVW * 4);
  // phase B aliases phase A region (only used after phase A dead)
  off = phase;
  int* qkey = (int*)alloc(BHS * 4);
  int* kkey = (int*)alloc(BHS * 4);
  int* qIdx = (int*)alloc(BHS * 4);
  int* kIdx = (int*)alloc(BHS * 4);
  f16* qs   = (f16*)alloc(BHS * DH_ * 2);
  f16* ks   = (f16*)alloc(BHS * DH_ * 2);
  f16* vsT  = (f16*)alloc(BHS * DH_ * 2);
  float* Xf = (float*)alloc((size_t)BS_ * DM * 4);
  float* Y  = (float*)alloc((size_t)BS_ * DM * 4);
  float* lseB = (float*)alloc(BHS * 4);
  f16* ksub  = (f16*)alloc((size_t)BH_ * SAMPLE_ * DH_ * 2);
  f16* vsubT = (f16*)alloc((size_t)BH_ * SAMPLE_ * DH_ * 2);

  // 0) pack weights -> f16 Bt[Npad][K] (zero-padded)
  k_packw<<<(QPROJ * DM) / 256, 256, 0, stream>>>(W_dq, Bt_dq, DM, QPROJ, 0);
  k_packw<<<(DM * QPROJ) / 256, 256, 0, stream>>>(W_uq, Bt_uq, QPROJ, DM, 0);
  k_packw<<<(CKVP * DM) / 256, 256, 0, stream>>>(W_dkv, Bt_dkv, DM, CKVW, 0);
  k_packw<<<(KVW * KVPROJ) / 256, 256, 0, stream>>>(W_ukv, Bt_ukv, KVPROJ, KVW, 0);
  k_packw<<<(DM * DM) / 256, 256, 0, stream>>>(W_o, Bt_o, DM, DM, 1);

  // 1) tmp1 = x @ W_dq
  k_gemm<<<dim3(BS_ / 128, QPROJ / 64), 256, 0, stream>>>(x, Bt_dq, tmp1, BS_, QPROJ, DM);
  // 2) cq = LN(tmp1)
  k_layernorm<<<BS_, 256, 0, stream>>>(tmp1, QPROJ, QPROJ, g_q, b_q, cq);
  // 3) Qbuf = cq @ W_uq
  k_gemm<<<dim3(BS_ / 128, DM / 64), 256, 0, stream>>>(cq, Bt_uq, Qbuf, BS_, DM, QPROJ);
  // 4) ckv = x @ W_dkv   (N padded to 576, pad cols produce zeros)
  k_gemm<<<dim3(BS_ / 128, CKVP / 64), 256, 0, stream>>>(x, Bt_dkv, ckv, BS_, CKVP, DM);
  // 5) kvl = LN(ckv[:, :512])
  k_layernorm<<<BS_, 256, 0, stream>>>(ckv, CKVP, KVPROJ, g_kv, b_kv, kvl);
  // 6) KVb = kvl @ W_ukv
  k_gemm<<<dim3(BS_ / 128, KVW / 64), 256, 0, stream>>>(kvl, Bt_ukv, KVb, BS_, KVW, KVPROJ);
  // 7) RoPE + split into q,k,v
  k_assemble<<<(B_ * S_ * HEADS_) / 256, 256, 0, stream>>>(Qbuf, KVb, ckv, qv, kv, vv);
  // 8) LSH keys
  k_lsh<<<(int)(BHS / 256), 256, 0, stream>>>(qv, pd, perm, qkey);
  k_lsh<<<(int)(BHS / 256), 256, 0, stream>>>(kv, pd, perm, kkey);
  // 9) stable sorts
  k_sort<<<BH_, 128, 0, stream>>>(qkey, qIdx);
  k_sort<<<BH_, 128, 0, stream>>>(kkey, kIdx);
  // 10) gather sorted (f16) + V transposed
  k_gather<<<(int)(BHS / 256), 256, 0, stream>>>(qv, kv, vv, qIdx, kIdx, qs, ks, vsT);
  // 11) sampled gathers
  k_gather_sampled<<<(B_ * HEADS_ * SAMPLE_) / 256, 256, 0, stream>>>(ks, vsT, samp, ksub, vsubT);
  // 12) block-diagonal attention
  k_blockattn<<<B_ * HEADS_ * NB_, 256, 0, stream>>>(qs, ks, vsT, Xf, lseB);
  // 13) Y = Xf @ W_o^T  (W_o packed as Bt directly)
  k_gemm<<<dim3(BS_ / 128, DM / 64), 256, 0, stream>>>(Xf, Bt_o, Y, BS_, DM, DM);
  // 14) sampled attention + blend + scatter to final output
  k_sampledattn<<<B_ * HEADS_ * NB_, 256, 0, stream>>>(qs, ksub, vsubT, samp, qIdx, lseB, Y, out);
}
